// LlamaAttention_20933670601350
// MI455X (gfx1250) — matrix operand order repaired and checked by reading, not compiled, not run on hardware
//
#include <hip/hip_runtime.h>
#include <hip/hip_bf16.h>

// ---------------------------------------------------------------------------
// Llama attention block for MI455X (gfx1250, wave32, WMMA + Tensor Data Mover).
// All GEMM-shaped math uses v_wmma_f32_16x16x32_bf16 (bf16 in, fp32 accum).
// Tile staging into LDS is done by the TDM (tensor_load_to_lds) with double
// buffering; RoPE / softmax / normalization kept in fp32 VALU.
// ---------------------------------------------------------------------------

typedef __attribute__((ext_vector_type(16))) __bf16 bf16x16;
typedef __attribute__((ext_vector_type(8)))  __bf16 bf16x8;
typedef __attribute__((ext_vector_type(8)))  float  f32x8;
typedef unsigned int u32x4 __attribute__((ext_vector_type(4)));
typedef int          i32x8 __attribute__((ext_vector_type(8)));
typedef int          i32x4 __attribute__((ext_vector_type(4)));

static constexpr int B_   = 2;
static constexpr int S_   = 2048;
static constexpr int HID_ = 4096;
static constexpr int NH_  = 32;
static constexpr int NKV_ = 8;
static constexpr int HD_  = 128;
static constexpr int G_   = NH_ / NKV_;      // 4
static constexpr int TOK_ = B_ * S_;         // 4096
static constexpr float ROPE_THETA_ = 10000.0f;

#if defined(__has_builtin)
#  if __has_builtin(__builtin_amdgcn_tensor_load_to_lds) && \
      __has_builtin(__builtin_amdgcn_s_wait_tensorcnt)
#    define HAVE_TDM 1
#  endif
#endif
#if defined(HAVE_TDM) && defined(__has_include)
#  if __has_include(<hip/amd_detail/amd_gfx1250_TDM.h>)
#    define TDM_6ARG 1   // therock/clang-23 toolchain: 6-arg builtin
#  endif
#endif

__device__ __forceinline__ f32x8 wmma_bf16(bf16x16 a, bf16x16 b, f32x8 c) {
  // (neg_a, A, neg_b, B, c_mod, C, reuse_a, reuse_b)
  return __builtin_amdgcn_wmma_f32_16x16x32_bf16(false, a, false, b, (short)0, c, false, false);
}

// Low 32 bits of the flat address of an LDS object = LDS byte offset
// (shared aperture occupies addr[63:32]; ISA section 10.2).
__device__ __forceinline__ unsigned int lds_off(const void* p) {
  return (unsigned int)(uintptr_t)p;
}

// TDM 2D tile load: bf16 elements (data_size=1 -> 2 bytes).
// Copies tile_h rows of tile_w elements, row stride row_stride (elements),
// from global gsrc into contiguous LDS at lds_byte_off.
__device__ __forceinline__ void tdm_load_2d_bf16(
    const void* gsrc, unsigned int lds_byte_off,
    unsigned int tile_w, unsigned int tile_h,
    unsigned int row_stride, unsigned int tensor_w, unsigned int tensor_h) {
#ifdef HAVE_TDM
  unsigned long long ga = (unsigned long long)(uintptr_t)gsrc;
  u32x4 g0;
  g0[0] = 1u;                                  // count=1 valid descriptor, user mode
  g0[1] = lds_byte_off;                        // lds_addr
  g0[2] = (unsigned int)ga;                    // global_addr[31:0]
  g0[3] = (unsigned int)((ga >> 32) & 0x1FFFFFFu) | (2u << 30);  // addr[56:32] | type=2
  i32x8 g1;
  g1[0] = (int)(1u << 16);                     // wg_mask=0, data_size=1 (2B), no flags
  g1[1] = (int)((tensor_w & 0xFFFFu) << 16);   // [31:16] = tensor_dim0[15:0]
  g1[2] = (int)((tensor_w >> 16) | ((tensor_h & 0xFFFFu) << 16));
  g1[3] = (int)((tensor_h >> 16) | (tile_w << 16));   // tile_dim0
  g1[4] = (int)(tile_h & 0xFFFFu);             // tile_dim1; tile_dim2=0 (2D)
  g1[5] = (int)row_stride;                     // tensor_dim0_stride[31:0]
  g1[6] = 0;                                   // stride0[47:32]=0, stride1 lo=0
  g1[7] = 0;
  i32x4 z4 = {0, 0, 0, 0};
#  ifdef TDM_6ARG
  i32x8 z8 = {0, 0, 0, 0, 0, 0, 0, 0};
  __builtin_amdgcn_tensor_load_to_lds(g0, g1, z4, z4, z8, 0);
#  else
  __builtin_amdgcn_tensor_load_to_lds(g0, g1, z4, z4, 0);
#  endif
#else
  (void)gsrc; (void)lds_byte_off; (void)tile_w; (void)tile_h;
  (void)row_stride; (void)tensor_w; (void)tensor_h;
#endif
}

// ---------------------------------------------------------------------------
// fp32 -> bf16 elementwise cast
// ---------------------------------------------------------------------------
__global__ void cast_to_bf16(const float* __restrict__ src, __bf16* __restrict__ dst, int n) {
  int i = blockIdx.x * blockDim.x + threadIdx.x;
  if (i < n) dst[i] = (__bf16)src[i];
}

// ---------------------------------------------------------------------------
// C[M,N] (fp32) = A[M,K] (bf16) * W[N,K]^T (bf16)   -- "NT" GEMM
// Block: 256 threads = 8 waves (2 M x 4 N), tile 128x128, K-step 32.
// Per wave: 64x32 output = 4x2 WMMA accumulators.
// Tiles staged by TDM into double-buffered LDS; wave 0 drives the DMA.
// ---------------------------------------------------------------------------
__global__ __launch_bounds__(256) void gemm_bf16_nt(
    const __bf16* __restrict__ A, const __bf16* __restrict__ W,
    float* __restrict__ C, int M, int N, int K) {
  __shared__ __bf16 Alds[2][128 * 32];   // 2 x 8 KB
  __shared__ __bf16 Wlds[2][128 * 32];   // 2 x 8 KB

  const int tid  = threadIdx.x;
  const int wave = tid >> 5;
  const int lane = tid & 31;
  const int hlf  = lane >> 4;        // lane half (0/1)
  const int l16  = lane & 15;
  const int wm   = wave >> 2;        // 0..1
  const int wn   = wave & 3;         // 0..3
  const int blockM = blockIdx.y * 128;
  const int blockN = blockIdx.x * 128;
  const int nsteps = K >> 5;

  f32x8 acc[4][2] = {};

#ifdef HAVE_TDM
  if (wave == 0) {   // uniform branch: only wave 0 issues the DMA
    tdm_load_2d_bf16(&A[(size_t)blockM * K], lds_off(&Alds[0][0]), 32, 128, K, K, M);
    tdm_load_2d_bf16(&W[(size_t)blockN * K], lds_off(&Wlds[0][0]), 32, 128, K, K, N);
  }
#endif

  for (int kk = 0; kk < nsteps; ++kk) {
    const int cur = kk & 1;
#ifdef HAVE_TDM
    if (wave == 0) __builtin_amdgcn_s_wait_tensorcnt(0);
    __syncthreads();                       // buffer `cur` ready for all waves
    if (wave == 0 && kk + 1 < nsteps) {    // prefetch next tile into other buffer
      const int k0n = (kk + 1) << 5;
      tdm_load_2d_bf16(&A[(size_t)blockM * K + k0n], lds_off(&Alds[cur ^ 1][0]), 32, 128, K, K, M);
      tdm_load_2d_bf16(&W[(size_t)blockN * K + k0n], lds_off(&Wlds[cur ^ 1][0]), 32, 128, K, K, N);
    }
#else
    const int k0 = kk << 5;
    __syncthreads();
#pragma unroll
    for (int p = 0; p < 2; ++p) {
      int idx = (p * 256 + tid) * 8;
      int r = idx >> 5, c = idx & 31;
      *(bf16x8*)&Alds[cur][idx] = *(const bf16x8*)&A[(size_t)(blockM + r) * K + k0 + c];
      *(bf16x8*)&Wlds[cur][idx] = *(const bf16x8*)&W[(size_t)(blockN + r) * K + k0 + c];
    }
    __syncthreads();
#endif

    // A fragments: lane l16 = row, K contiguous 16 per lane half: {0..15} vs {16..31}.
    bf16x16 afrag[4];
#pragma unroll
    for (int mi = 0; mi < 4; ++mi) {
      const __bf16* src = &Alds[cur][(wm * 64 + mi * 16 + l16) * 32 + hlf * 16];
      bf16x8 lo = *(const bf16x8*)src;
      bf16x8 hi = *(const bf16x8*)(src + 8);
#pragma unroll
      for (int i = 0; i < 8; ++i) { afrag[mi][i] = lo[i]; afrag[mi][8 + i] = hi[i]; }
    }
#pragma unroll
    for (int ni = 0; ni < 2; ++ni) {
      // B fragment: col n = l16 (row of W), K contiguous 16 per lane half.
      bf16x16 bfrag = *(const bf16x16*)&Wlds[cur][(wn * 32 + ni * 16 + l16) * 32 + hlf * 16];
#pragma unroll
      for (int mi = 0; mi < 4; ++mi)
        acc[mi][ni] = wmma_bf16(afrag[mi], bfrag, acc[mi][ni]);
    }
#ifdef HAVE_TDM
    __syncthreads();   // all reads of buffer `cur` done before it is refilled
#endif
  }

  // Store C: vgpr r -> row (r + 8*half), lane l16 -> col.
#pragma unroll
  for (int mi = 0; mi < 4; ++mi)
#pragma unroll
    for (int ni = 0; ni < 2; ++ni) {
      int rbase = blockM + wm * 64 + mi * 16 + hlf * 8;
      int col   = blockN + wn * 32 + ni * 16 + l16;
#pragma unroll
      for (int r = 0; r < 8; ++r)
        C[(size_t)(rbase + r) * N + col] = acc[mi][ni][r];
    }
}

// ---------------------------------------------------------------------------
// RoPE + cast + transpose: X fp32 [B*S, nH*HD] -> Out bf16 [B, nH, S, HD].
// ---------------------------------------------------------------------------
__global__ void rope_cast_transpose(const float* __restrict__ X, __bf16* __restrict__ Out,
                                    int nHeads, int total) {
  int idx = blockIdx.x * blockDim.x + threadIdx.x;
  if (idx >= total) return;
  int i   = idx & 63;                  // frequency index 0..63
  int t2  = idx >> 6;
  int hh  = t2 % nHeads;
  int tok = t2 / nHeads;               // 0..B*S-1
  int s   = tok & (S_ - 1);
  int b   = tok / S_;

  const float* src = X + (size_t)tok * (nHeads * HD_) + hh * HD_;
  float x1 = src[i];
  float x2 = src[i + 64];
  float freq = (float)s * __powf(ROPE_THETA_, -(float)i * (1.0f / 64.0f));
  float sn, cs;
  __sincosf(freq, &sn, &cs);
  __bf16* dst = Out + (((size_t)b * nHeads + hh) * S_ + s) * HD_;
  dst[i]      = (__bf16)(x1 * cs - x2 * sn);
  dst[i + 64] = (__bf16)(x2 * cs + x1 * sn);
}

// ---------------------------------------------------------------------------
// V cast + transpose: fp32 [B*S, NKV*HD] -> bf16 [B, NKV, S, HD]
// ---------------------------------------------------------------------------
__global__ void cast_transpose_v(const float* __restrict__ X, __bf16* __restrict__ Out, int total) {
  int idx = blockIdx.x * blockDim.x + threadIdx.x;
  if (idx >= total) return;
  int d   = idx & (HD_ - 1);
  int t2  = idx >> 7;
  int kh  = t2 % NKV_;
  int tok = t2 / NKV_;
  int s   = tok & (S_ - 1);
  int b   = tok / S_;
  Out[(((size_t)b * NKV_ + kh) * S_ + s) * HD_ + d] =
      (__bf16)X[(size_t)tok * (NKV_ * HD_) + kh * HD_ + d];
}

// ---------------------------------------------------------------------------
// Flash attention (causal, GQA). Grid: (S/64, B*NH). Block: 128 thr = 4 waves.
// Each wave owns a 16-query tile; streams keys in 32-wide blocks.
// K blocks staged by TDM; V staged manually (transposed in LDS).
// ---------------------------------------------------------------------------
__global__ __launch_bounds__(128) void flash_attn(
    const __bf16* __restrict__ Q,   // [B, NH, S, HD]
    const __bf16* __restrict__ Kt,  // [B, NKV, S, HD]
    const __bf16* __restrict__ Vt,  // [B, NKV, S, HD]
    __bf16* __restrict__ Ctx) {     // [B, S, NH*HD]
  __shared__ __bf16 Klds[32 * 128];     // 8 KB, [key][dim]
  __shared__ __bf16 VldsT[128 * 32];    // 8 KB, [dim][key]
  __shared__ __bf16 Plds[4][16 * 32];   // 4 KB, per-wave P relayout buffer

  const int tid  = threadIdx.x;
  const int wave = tid >> 5;
  const int lane = tid & 31;
  const int hlf  = lane >> 4;
  const int l16  = lane & 15;

  const int bh = blockIdx.y;
  const int b  = bh / NH_;
  const int h  = bh % NH_;
  const int kh = h / G_;
  const int qBase = blockIdx.x * 64;

  const __bf16* Qp = Q  + (((size_t)b * NH_  + h ) * S_) * HD_;
  const __bf16* Kp = Kt + (((size_t)b * NKV_ + kh) * S_) * HD_;
  const __bf16* Vp = Vt + (((size_t)b * NKV_ + kh) * S_) * HD_;

  // Load Q A-fragments for this wave's 16 rows (kept in registers).
  bf16x16 qf[4];
  {
    const int qrowA = qBase + wave * 16 + l16;   // A-layout: row = lane&15
#pragma unroll
    for (int c = 0; c < 4; ++c) {
      const __bf16* src = Qp + (size_t)qrowA * HD_ + c * 32 + hlf * 16;
      bf16x8 lo = *(const bf16x8*)src;
      bf16x8 hi = *(const bf16x8*)(src + 8);
#pragma unroll
      for (int i = 0; i < 8; ++i) { qf[c][i] = lo[i]; qf[c][8 + i] = hi[i]; }
    }
  }

  f32x8 acc[8] = {};
  float rm[8], rl[8];
#pragma unroll
  for (int r = 0; r < 8; ++r) { rm[r] = -1e30f; rl[r] = 0.0f; }

  const float scale = 0.08838834764831845f;   // 1/sqrt(128)
  const int nblk = (qBase + 64) / 32;         // causal key-block bound (uniform per WG)

  for (int j = 0; j < nblk; ++j) {
    const int kBase = j * 32;
    __syncthreads();   // previous iteration's LDS reads complete
#ifdef HAVE_TDM
    if (wave == 0)     // 32 keys x 128 dims, contiguous rows
      tdm_load_2d_bf16(Kp + (size_t)kBase * HD_, lds_off(Klds), 128, 32, 128, 128, S_);
#endif
    // Stage V transposed (and K too if no TDM).
#pragma unroll
    for (int p = 0; p < 4; ++p) {
      int idx = (p * 128 + tid) * 8;           // 0..4095
#ifndef HAVE_TDM
      *(bf16x8*)&Klds[idx] = *(const bf16x8*)&Kp[(size_t)kBase * HD_ + idx];
#endif
      int key = idx >> 7, d0 = idx & 127;
      bf16x8 v = *(const bf16x8*)&Vp[(size_t)(kBase + key) * HD_ + d0];
#pragma unroll
      for (int i = 0; i < 8; ++i) VldsT[(d0 + i) * 32 + key] = v[i];
    }
#ifdef HAVE_TDM
    if (wave == 0) __builtin_amdgcn_s_wait_tensorcnt(0);
#endif
    __syncthreads();

    // S = Q @ K^T : two 16-key tiles, K-dim 128 = 4 WMMAs each.
    f32x8 sc[2] = {};
#pragma unroll
    for (int t = 0; t < 2; ++t) {
      const int key = t * 16 + l16;            // B-layout: col = lane&15
#pragma unroll
      for (int c = 0; c < 4; ++c) {
        bf16x16 bf = *(const bf16x16*)&Klds[key * 128 + c * 32 + hlf * 16];
        sc[t] = wmma_bf16(qf[c], bf, sc[t]);
      }
    }

    // Mask + online softmax (row = r + 8*half in C layout).
#pragma unroll
    for (int r = 0; r < 8; ++r) {
      int qrow = qBase + wave * 16 + r + hlf * 8;
      float s0 = sc[0][r] * scale;
      float s1 = sc[1][r] * scale;
      if (kBase + l16 > qrow)      s0 = -1e30f;
      if (kBase + 16 + l16 > qrow) s1 = -1e30f;
      float mx = fmaxf(s0, s1);
#pragma unroll
      for (int off = 1; off < 16; off <<= 1) mx = fmaxf(mx, __shfl_xor(mx, off, 32));
      float mnew = fmaxf(rm[r], mx);
      float p0 = __expf(s0 - mnew);
      float p1 = __expf(s1 - mnew);
      float rs = p0 + p1;
#pragma unroll
      for (int off = 1; off < 16; off <<= 1) rs += __shfl_xor(rs, off, 32);
      float alpha = __expf(rm[r] - mnew);
      rl[r] = rl[r] * alpha + rs;
      rm[r] = mnew;
#pragma unroll
      for (int nt = 0; nt < 8; ++nt) acc[nt][r] *= alpha;
      // Spill P to per-wave LDS to re-stripe C-layout -> A-layout.
      int prow = r + hlf * 8;
      Plds[wave][prow * 32 + l16]      = (__bf16)p0;
      Plds[wave][prow * 32 + 16 + l16] = (__bf16)p1;
    }

    // Reload P as A fragment (same-wave DS ops are in-order: RAW safe).
    bf16x16 pf;
    {
      const __bf16* src = &Plds[wave][l16 * 32 + hlf * 16];
      bf16x8 lo = *(const bf16x8*)src;
      bf16x8 hi = *(const bf16x8*)(src + 8);
#pragma unroll
      for (int i = 0; i < 8; ++i) { pf[i] = lo[i]; pf[8 + i] = hi[i]; }
    }

    // O += P @ V : 8 dim tiles of 16.
#pragma unroll
    for (int nt = 0; nt < 8; ++nt) {
      bf16x16 vf = *(const bf16x16*)&VldsT[(nt * 16 + l16) * 32 + hlf * 16];
      acc[nt] = wmma_bf16(pf, vf, acc[nt]);
    }
  }

  // Normalize and write context in [B, S, NH*HD] bf16.
#pragma unroll
  for (int nt = 0; nt < 8; ++nt)
#pragma unroll
    for (int r = 0; r < 8; ++r) {
      int qrow = qBase + wave * 16 + r + hlf * 8;
      float o = acc[nt][r] / rl[r];
      Ctx[((size_t)b * S_ + qrow) * HID_ + h * HD_ + nt * 16 + l16] = (__bf16)o;
    }
}

// ---------------------------------------------------------------------------
// Host-side orchestration
// ---------------------------------------------------------------------------
extern "C" void kernel_launch(void* const* d_in, const int* in_sizes, int n_in,
                              void* d_out, int out_size, void* d_ws, size_t ws_size,
                              hipStream_t stream) {
  (void)in_sizes; (void)n_in; (void)out_size; (void)ws_size;
  const float* h_f32  = (const float*)d_in[0];   // [B,S,HID]
  const float* Wq_f32 = (const float*)d_in[1];   // [4096,4096]
  const float* Wk_f32 = (const float*)d_in[2];   // [1024,4096]
  const float* Wv_f32 = (const float*)d_in[3];   // [1024,4096]
  const float* Wo_f32 = (const float*)d_in[4];   // [4096,4096]
  float* out = (float*)d_out;

  char* ws = (char*)d_ws;
  size_t off = 0;
  auto carve = [&](size_t bytes) { char* p = ws + off; off += (bytes + 255) & ~(size_t)255; return p; };

  __bf16* hbf  = (__bf16*)carve((size_t)TOK_ * HID_ * 2);
  __bf16* wqbf = (__bf16*)carve((size_t)HID_ * HID_ * 2);
  __bf16* wkbf = (__bf16*)carve((size_t)(NKV_ * HD_) * HID_ * 2);
  __bf16* wvbf = (__bf16*)carve((size_t)(NKV_ * HD_) * HID_ * 2);
  __bf16* wobf = (__bf16*)carve((size_t)HID_ * HID_ * 2);
  __bf16* Qr   = (__bf16*)carve((size_t)TOK_ * NH_  * HD_ * 2);
  __bf16* Kr   = (__bf16*)carve((size_t)TOK_ * NKV_ * HD_ * 2);
  __bf16* Vr   = (__bf16*)carve((size_t)TOK_ * NKV_ * HD_ * 2);
  __bf16* Ctx  = (__bf16*)carve((size_t)TOK_ * HID_ * 2);
  float*  tmp  = (float*) carve((size_t)TOK_ * HID_ * 4);   // largest fp32 temp

  const int CT = 256;
  auto cast = [&](const float* s, __bf16* d, int n) {
    cast_to_bf16<<<(n + CT - 1) / CT, CT, 0, stream>>>(s, d, n);
  };
  cast(h_f32,  hbf,  TOK_ * HID_);
  cast(Wq_f32, wqbf, HID_ * HID_);
  cast(Wk_f32, wkbf, NKV_ * HD_ * HID_);
  cast(Wv_f32, wvbf, NKV_ * HD_ * HID_);
  cast(Wo_f32, wobf, HID_ * HID_);

  dim3 gq(HID_ / 128, TOK_ / 128);           // (32, 32)
  dim3 gkv((NKV_ * HD_) / 128, TOK_ / 128);  // (8, 32)

  // Q projection -> RoPE -> [B,NH,S,HD] bf16
  gemm_bf16_nt<<<gq, 256, 0, stream>>>(hbf, wqbf, tmp, TOK_, HID_, HID_);
  {
    int total = TOK_ * NH_ * 64;
    rope_cast_transpose<<<(total + CT - 1) / CT, CT, 0, stream>>>(tmp, Qr, NH_, total);
  }
  // K projection -> RoPE
  gemm_bf16_nt<<<gkv, 256, 0, stream>>>(hbf, wkbf, tmp, TOK_, NKV_ * HD_, HID_);
  {
    int total = TOK_ * NKV_ * 64;
    rope_cast_transpose<<<(total + CT - 1) / CT, CT, 0, stream>>>(tmp, Kr, NKV_, total);
  }
  // V projection -> cast/transpose
  gemm_bf16_nt<<<gkv, 256, 0, stream>>>(hbf, wvbf, tmp, TOK_, NKV_ * HD_, HID_);
  {
    int total = TOK_ * NKV_ * HD_;
    cast_transpose_v<<<(total + CT - 1) / CT, CT, 0, stream>>>(tmp, Vr, total);
  }
  // Attention
  flash_attn<<<dim3(S_ / 64, B_ * NH_), 128, 0, stream>>>(Qr, Kr, Vr, Ctx);
  // Output projection -> fp32 d_out
  gemm_bf16_nt<<<gq, 256, 0, stream>>>(Ctx, wobf, out, TOK_, HID_, HID_);
}
